// AttentionCompute_72868415143979
// MI455X (gfx1250) — compile-verified, matrix operand order checked by reading, not run on hardware
//
#include <hip/hip_runtime.h>

// ---------------------------------------------------------------------------
// Flash attention, B=2 H=16 S=2048 D=64, fp32 in/out, f16 WMMA compute.
// gfx1250 (CDNA5) wave32: V_WMMA_F32_16X16X32_F16
//  - 4 waves/block, 16 q-rows per wave (64 q-rows per block)
//  - 64-key tiles, double-buffered global->LDS staging
//  - row-sum of P computed by WMMA against an all-ones B operand
// ---------------------------------------------------------------------------

typedef __attribute__((ext_vector_type(16))) _Float16 v16h;
typedef __attribute__((ext_vector_type(8)))  _Float16 v8h;
typedef __attribute__((ext_vector_type(8)))  float    v8f;

#define S_LEN   2048
#define DHEAD   64
#define KT      64            // keys per tile
#define NTILES  (S_LEN / KT)  // 32
#define KSTR    72            // K lds row stride in halves (144B: 16B-aligned, 36-bank step)
#define VSTR    72            // Vt lds row stride in halves
#define PSTR    72            // P lds row stride in halves
#define WAVES   4

static __device__ inline v8f wmma_f16(v16h a, v16h b, v8f c) {
    return __builtin_amdgcn_wmma_f32_16x16x32_f16(
        false, a, false, b, (short)0, c, false, false);
}

// 16x32 "A layout" operand from an LDS row pointer.
// lane (lo,hi): halves[0..7] = row[hi*8 + 0..7], halves[8..15] = row[16 + hi*8 + 0..7]
static __device__ inline v16h make_a(const _Float16* rowp, int hi) {
    v8h a = *(const v8h*)(rowp + hi * 8);
    v8h b = *(const v8h*)(rowp + 16 + hi * 8);
    v16h r;
#pragma unroll
    for (int i = 0; i < 8; ++i) { r[i] = a[i]; r[i + 8] = b[i]; }
    return r;
}

__global__ __launch_bounds__(128) void
attn_fwd_kernel(const float* __restrict__ Q, const float* __restrict__ K,
                const float* __restrict__ V, float* __restrict__ O)
{
    __shared__ _Float16 KshA[KT * KSTR], KshB[KT * KSTR];      // 64 x 72 halves each
    __shared__ _Float16 VtsA[DHEAD * VSTR], VtsB[DHEAD * VSTR];// 64 x 72 (V transposed)
    __shared__ _Float16 Psh[WAVES * 16 * PSTR];                // per-wave P scratch

    const int tid  = threadIdx.x;
    const int wave = tid >> 5;
    const int lane = tid & 31;
    const int lo   = lane & 15;
    const int hi   = lane >> 4;

    const int bh    = blockIdx.x >> 5;   // (b*H+h); 32 q-blocks of 64 rows per head
    const int qblk  = blockIdx.x & 31;
    const int qrow0 = qblk * 64 + wave * 16;

    const float* Qb = Q + (size_t)bh * S_LEN * DHEAD;
    const float* Kb = K + (size_t)bh * S_LEN * DHEAD;
    const float* Vb = V + (size_t)bh * S_LEN * DHEAD;
    float*       Ob = O + (size_t)bh * S_LEN * DHEAD;

    // fold softmax scale (1/sqrt(64)) and log2(e) into Q so P = exp2(S - m)
    const float qscale = 0.125f * 1.44269504088896340736f;

    // ---- Q tile (16x64) as two A-layout operands (d 0..31 / 32..63)
    v16h qa0, qa1;
    {
        const float* qrow = Qb + (size_t)(qrow0 + lo) * DHEAD;
        const int c0 = hi * 8;
#pragma unroll
        for (int i = 0; i < 8; ++i) {
            qa0[i]     = (_Float16)(qrow[c0 + i]      * qscale);
            qa0[i + 8] = (_Float16)(qrow[16 + c0 + i] * qscale);
            qa1[i]     = (_Float16)(qrow[32 + c0 + i] * qscale);
            qa1[i + 8] = (_Float16)(qrow[48 + c0 + i] * qscale);
        }
    }

    v16h ones;
#pragma unroll
    for (int i = 0; i < 16; ++i) ones[i] = (_Float16)1.0f;

    // ---- accumulators: O (16x64 over 4 N-chunks), l (row sums), m (row max)
    v8f acc[4], lacc;
    float mrow[8];
#pragma unroll
    for (int n = 0; n < 4; ++n)
#pragma unroll
        for (int g = 0; g < 8; ++g) acc[n][g] = 0.0f;
#pragma unroll
    for (int g = 0; g < 8; ++g) { lacc[g] = 0.0f; mrow[g] = -1e30f; }

    _Float16* pw = &Psh[wave * 16 * PSTR];

    // cooperative stage of one 64-key tile: K row-major f16, V transposed f16
    auto stage = [&](int tile, _Float16* kdst, _Float16* vdst) {
        const int kb = tile * KT;
#pragma unroll
        for (int i = 0; i < 8; ++i) {
            const int slot = i * 128 + tid;    // 0..1023 float4 slots
            const int row  = slot >> 4;        // 0..63 key in tile
            const int col  = (slot & 15) << 2; // 0..60 d
            const float4 kd = *(const float4*)(Kb + (size_t)(kb + row) * DHEAD + col);
            const float4 vd = *(const float4*)(Vb + (size_t)(kb + row) * DHEAD + col);
            _Float16* kp = kdst + row * KSTR + col;
            kp[0] = (_Float16)kd.x; kp[1] = (_Float16)kd.y;
            kp[2] = (_Float16)kd.z; kp[3] = (_Float16)kd.w;
            vdst[(col + 0) * VSTR + row] = (_Float16)vd.x;
            vdst[(col + 1) * VSTR + row] = (_Float16)vd.y;
            vdst[(col + 2) * VSTR + row] = (_Float16)vd.z;
            vdst[(col + 3) * VSTR + row] = (_Float16)vd.w;
            if (tile + 1 < NTILES) {          // L2 prefetch (global_prefetch_b8)
                __builtin_prefetch(Kb + (size_t)(kb + KT + row) * DHEAD + col, 0, 1);
                __builtin_prefetch(Vb + (size_t)(kb + KT + row) * DHEAD + col, 0, 1);
            }
        }
    };

    stage(0, KshA, VtsA);

    for (int t = 0; t < NTILES; ++t) {
        __syncthreads();   // staged tile t visible; prior reads of the other buffer done
        const int cur = t & 1;
        const _Float16* kbuf = cur ? KshB : KshA;
        const _Float16* vbuf = cur ? VtsB : VtsA;
        if (t + 1 < NTILES)
            stage(t + 1, cur ? KshA : KshB, cur ? VtsA : VtsB);

        // ---- scores S = Q*K^T (16 x 64), four 16-key C tiles
        v8f s[4];
#pragma unroll
        for (int st = 0; st < 4; ++st) {
#pragma unroll
            for (int g = 0; g < 8; ++g) s[st][g] = 0.0f;
            const _Float16* kr = kbuf + (st * 16 + lo) * KSTR;
            s[st] = wmma_f16(qa0, make_a(kr, hi),      s[st]);
            s[st] = wmma_f16(qa1, make_a(kr + 32, hi), s[st]);
        }

        // ---- row max across 64 keys (intra-lane over 4 tiles, then 16 lanes)
        float mt[8];
#pragma unroll
        for (int g = 0; g < 8; ++g)
            mt[g] = fmaxf(fmaxf(s[0][g], s[1][g]), fmaxf(s[2][g], s[3][g]));
#pragma unroll
        for (int msk = 1; msk < 16; msk <<= 1)
#pragma unroll
            for (int g = 0; g < 8; ++g)
                mt[g] = fmaxf(mt[g], __shfl_xor(mt[g], msk, 32));

        float corr[8];
#pragma unroll
        for (int g = 0; g < 8; ++g) {
            const float mn = fmaxf(mrow[g], mt[g]);
            corr[g] = exp2f(mrow[g] - mn);
            mrow[g] = mn;
        }
#pragma unroll
        for (int st = 0; st < 4; ++st)
#pragma unroll
            for (int g = 0; g < 8; ++g)
                s[st][g] = exp2f(s[st][g] - mrow[g]);

        // rescale running accumulators (O and l) by corr
#pragma unroll
        for (int n = 0; n < 4; ++n)
#pragma unroll
            for (int g = 0; g < 8; ++g) acc[n][g] *= corr[g];
#pragma unroll
        for (int g = 0; g < 8; ++g) lacc[g] *= corr[g];

        // ---- P: C-layout -> per-wave LDS scratch -> two A-layout operands
#pragma unroll
        for (int st = 0; st < 4; ++st)
#pragma unroll
            for (int g = 0; g < 8; ++g)
                pw[(g + 8 * hi) * PSTR + st * 16 + lo] = (_Float16)s[st][g];
        const v16h pa0 = make_a(pw + lo * PSTR, hi);       // keys 0..31
        const v16h pa1 = make_a(pw + lo * PSTR + 32, hi);  // keys 32..63

        // ---- l += rowsum(P) via WMMA against all-ones B (replicated per column)
        lacc = wmma_f16(pa0, ones, lacc);
        lacc = wmma_f16(pa1, ones, lacc);

        // ---- O += P * V  (B operands = rows of V^T in LDS)
#pragma unroll
        for (int n = 0; n < 4; ++n) {
            const _Float16* vr = vbuf + (n * 16 + lo) * VSTR;
            acc[n] = wmma_f16(pa0, make_a(vr, hi),      acc[n]);
            acc[n] = wmma_f16(pa1, make_a(vr + 32, hi), acc[n]);
        }
    }

    // ---- normalize and write fp32 output
#pragma unroll
    for (int g = 0; g < 8; ++g) {
        const int r = qrow0 + g + 8 * hi;
        const float inv = 1.0f / lacc[g];
        float* orow = Ob + (size_t)r * DHEAD;
#pragma unroll
        for (int n = 0; n < 4; ++n)
            orow[n * 16 + lo] = acc[n][g] * inv;
    }
}

extern "C" void kernel_launch(void* const* d_in, const int* in_sizes, int n_in,
                              void* d_out, int out_size, void* d_ws, size_t ws_size,
                              hipStream_t stream) {
    (void)in_sizes; (void)n_in; (void)out_size; (void)d_ws; (void)ws_size;
    const float* q = (const float*)d_in[0];
    const float* k = (const float*)d_in[1];
    const float* v = (const float*)d_in[2];
    float* o = (float*)d_out;
    dim3 grid(1024), block(128);   // (B*H)*(S/64) blocks, 4 waves each
    attn_fwd_kernel<<<grid, block, 0, stream>>>(q, k, v, o);
}
